// QuestionReader_60473139528395
// MI455X (gfx1250) — matrix operand order repaired and checked by reading, not compiled, not run on hardware
//
#include <hip/hip_runtime.h>

// ---------------------------------------------------------------------------
// RWA-RNN for MI455X (gfx1250): TDM (tensor_load_to_lds) stages activation
// tiles, bf16 WMMA for the time-parallel GEMM and the per-step h@W recurrence
// GEMMs; f32 state/transcendentals.
// ---------------------------------------------------------------------------

typedef __attribute__((ext_vector_type(16))) __bf16 v16bf;
typedef __attribute__((ext_vector_type(8)))  float  v8f;
typedef __attribute__((ext_vector_type(4)))  float  f32x4;
typedef __attribute__((ext_vector_type(2)))  float  f32x2;
typedef __attribute__((ext_vector_type(4)))  unsigned int u32x4;
typedef __attribute__((ext_vector_type(8)))  unsigned int u32x8;
typedef unsigned int u32;
typedef unsigned long long u64;

#define D_IN   300
#define KPAD   320
#define NC     192          // fused output cols: [u | g_x | a_x]
#define HIDD   64
#define BATCH  512
#define SEQT   128
#define BT     65536        // BATCH * SEQT

union Frag16 {
    v16bf v;
    u32   u[8];
    f32x4 f4[2];
};

__device__ __forceinline__ int a_dw(int j, int h) {
    // packed-dword index within a 16x32 bf16 A-matrix row for frag element pair j
    // lane half h: half 0 holds K {0..15}, half 1 holds K {16..31}
    return j + 8 * h;
}

__device__ __forceinline__ v8f v8f_zero() {
    v8f r;
#pragma unroll
    for (int i = 0; i < 8; ++i) r[i] = 0.0f;
    return r;
}

__device__ __forceinline__ u32 pack_bf16(float a, float b) {
    union { __bf16 h[2]; u32 u; } c;
    c.h[0] = (__bf16)a;
    c.h[1] = (__bf16)b;
    return c.u;
}

__device__ __forceinline__ float tanh_fast(float x) {
    // monotone, saturates correctly for |x| large
    return 1.0f - 2.0f / (__expf(2.0f * x) + 1.0f);
}

// ---------------------------------------------------------------------------
// Kernel P: pack weights -> bf16, N-major/K-minor layouts; fold biases.
//   WcT  [192][320] bf16 : col n<64 -> Wu, 64..127 -> Wg[:300], 128..191 -> Wa[:300]
//   WghT [64][64]  bf16 : WghT[n][k] = Wg[300+k][n]
//   WahT [64][64]  bf16 : WahT[n][k] = Wa[300+k][n]
//   bias [192] f32      : [bu | bg | 0]
// ---------------------------------------------------------------------------
__global__ __launch_bounds__(256) void rwa_pack_kernel(
    const float* __restrict__ Wu, const float* __restrict__ bu,
    const float* __restrict__ Wg, const float* __restrict__ bg,
    const float* __restrict__ Wa,
    __bf16* __restrict__ WcT, __bf16* __restrict__ WghT,
    __bf16* __restrict__ WahT, float* __restrict__ bias)
{
    int idx = blockIdx.x * 256 + threadIdx.x;
    if (idx < NC * KPAD) {
        int n = idx / KPAD, k = idx % KPAD;
        float v = 0.0f;
        if (k < D_IN) {
            if (n < 64)       v = Wu[k * 64 + n];
            else if (n < 128) v = Wg[k * 64 + (n - 64)];
            else              v = Wa[k * 64 + (n - 128)];
        }
        WcT[n * KPAD + k] = (__bf16)v;
    } else if (idx < NC * KPAD + 4096) {
        int i = idx - NC * KPAD;
        int n = i >> 6, k = i & 63;
        WghT[n * 64 + k] = (__bf16)Wg[(D_IN + k) * 64 + n];
    } else if (idx < NC * KPAD + 8192) {
        int i = idx - NC * KPAD - 4096;
        int n = i >> 6, k = i & 63;
        WahT[n * 64 + k] = (__bf16)Wa[(D_IN + k) * 64 + n];
    } else if (idx < NC * KPAD + 8192 + NC) {
        int c = idx - NC * KPAD - 8192;
        float v = 0.0f;
        if (c < 64)       v = bu[c];
        else if (c < 128) v = bg[c - 64];
        bias[c] = v;
    }
}

// ---------------------------------------------------------------------------
// Kernel G: fused GEMM  Out[2*BT, 192] = concat(in1,in2)[*,300] @ WcT^T + bias
// Block = 256 threads (8 waves), 128-row x 192-col tile per block.
// Staging: each wave issues ONE tensor_load_to_lds (TDM) for its 16x300 f32
// strip (16*1200 B), waits on TENSORcnt, then the whole K loop runs with no
// barriers: A frags are built from f32 LDS with v_cvt_pk_bf16_f32, B frags
// come from the N-major bf16 weight copy (L2-resident).
// Dynamic LDS: 128*300*4 = 153600 B (2 blocks / 320KB WGP).
// ---------------------------------------------------------------------------
__global__ __launch_bounds__(256) void rwa_gemm_kernel(
    const float* __restrict__ in1, const float* __restrict__ in2,
    const __bf16* __restrict__ WcT, const float* __restrict__ bias,
    float* __restrict__ Out)
{
    extern __shared__ float AsF[];   // [128][300] f32 activation tile

    const int tid  = threadIdx.x;
    const int lane = tid & 31;
    const int h    = lane >> 4;
    const int ln   = lane & 15;
    const int wave = __builtin_amdgcn_readfirstlane(tid >> 5);   // uniform

    const long r0 = (long)blockIdx.x * 128;
    const float* __restrict__ x = (r0 < BT) ? in1 : in2;
    const long xr0 = r0 & (BT - 1);

    // ---- TDM: DMA this wave's 16x300 f32 strip into LDS ----
    {
        const u64 ga = (u64)(const void*)(x + (xr0 + (long)wave * 16) * D_IN);
        const u32 lds_base =
            (u32)(unsigned long long)(const void*)AsF + (u32)(wave * 16 * D_IN * 4);

        u32x4 g0;
        g0[0] = 1u;                                   // count=1 (valid), user mode
        g0[1] = lds_base;                             // lds_addr (bytes)
        g0[2] = (u32)ga;                              // global_addr[31:0]
        g0[3] = ((u32)(ga >> 32) & 0x01FFFFFFu)       // global_addr[56:32]
                | 0x80000000u;                        // type=2 ("image")

        u32x8 g1;
        g1[0] = 2u << 16;                             // data_size=4B; no multicast/pad
        g1[1] = (u32)D_IN << 16;                      // tensor_dim0[15:0]=300
        g1[2] = 16u << 16;                            // tensor_dim0[31:16]=0, tensor_dim1=16
        g1[3] = (u32)D_IN << 16;                      // tile_dim0=300
        g1[4] = 16u;                                  // tile_dim1=16, tile_dim2=0
        g1[5] = (u32)D_IN;                            // tensor_dim0_stride=300
        g1[6] = 0u;                                   // stride hi, dim1_stride lo
        g1[7] = 0u;                                   // dim1_stride hi

        // groups 2/3 = NULL -> 2D tensor form
        asm volatile("tensor_load_to_lds %0, %1" :: "s"(g0), "s"(g1) : "memory");
        __builtin_amdgcn_s_wait_tensorcnt(0);
    }
    __syncthreads();

    v8f acc[12];
#pragma unroll
    for (int i = 0; i < 12; ++i) acc[i] = v8f_zero();

    const float* __restrict__ Ar = AsF + (wave * 16 + ln) * D_IN;

    for (int kc = 0; kc < 10; ++kc) {
        const int k0 = kc * 32;

        // --- A fragment (16x32 bf16) from the f32 LDS strip ---
        Frag16 af;
        if (k0 + 32 <= D_IN) {
#pragma unroll
            for (int j = 0; j < 8; ++j) {
                const f32x2 p = *(const f32x2*)(Ar + k0 + 2 * a_dw(j, h));
                af.u[j] = pack_bf16(p.x, p.y);
            }
        } else {   // last chunk: K in [288,320), zero-fill K >= 300
#pragma unroll
            for (int j = 0; j < 8; ++j) {
                const int kb = k0 + 2 * a_dw(j, h);
                const float x0 = (kb     < D_IN) ? Ar[kb]     : 0.0f;
                const float x1 = (kb + 1 < D_IN) ? Ar[kb + 1] : 0.0f;
                af.u[j] = pack_bf16(x0, x1);
            }
        }

        // --- 12 N-tiles: B frag (32x16, K-contiguous in WcT) + WMMA ---
#pragma unroll
        for (int nt = 0; nt < 12; ++nt) {
            Frag16 bf;
            const f32x4* bp =
                (const f32x4*)(WcT + (nt * 16 + ln) * KPAD + k0 + 16 * h);
            bf.f4[0] = bp[0];
            bf.f4[1] = bp[1];
            acc[nt] = __builtin_amdgcn_wmma_f32_16x16x32_bf16(
                false, af.v, false, bf.v, (short)0, acc[nt], false, false);
        }
    }

    // --- epilogue: D layout row = v + 8h, col = ln; add folded bias ---
    const long orow0 = r0 + wave * 16;
#pragma unroll
    for (int nt = 0; nt < 12; ++nt) {
        const int c  = nt * 16 + ln;
        const float bv = bias[c];
#pragma unroll
        for (int v = 0; v < 8; ++v) {
            const long m = orow0 + v + 8 * h;
            Out[m * NC + c] = acc[nt][v] + bv;
        }
    }
}

// ---------------------------------------------------------------------------
// Kernel R: the sequential scan. One 32-thread wave per block handles a
// 16-row batch strip for one input; grid = 2 inputs x 32 strips = 64 blocks.
// Per step: Gh = h@Wgh, Ah = h@Wah via 16 bf16 WMMAs (8 tiles x K=64),
// then the numerically-stable softmax-average update in f32. n/d/amax live
// in registers (the D-matrix layout pins each (b,h) cell to one lane).
// ---------------------------------------------------------------------------
__global__ __launch_bounds__(32) void rwa_scan_kernel(
    const float* __restrict__ Out, const __bf16* __restrict__ WghT,
    const __bf16* __restrict__ WahT, const float* __restrict__ s,
    float* __restrict__ hn)
{
    __shared__ __bf16 hA[16 * 66];     // h as 16x64 bf16 A-matrix, padded rows
    __shared__ __bf16 WgS[64 * 72];    // N-major/K-minor, padded (bank-safe)
    __shared__ __bf16 WaS[64 * 72];
    __shared__ float  xrow[16 * 196];  // staged [u|gx|ax] rows, padded

    const int lane  = threadIdx.x;
    const int h     = lane >> 4;
    const int ln    = lane & 15;
    const int input = blockIdx.x >> 5;
    const int b0    = (blockIdx.x & 31) * 16;

    // stage recurrence weights into LDS
    for (int i = lane; i < 64 * 64; i += 32) {
        const int n = i >> 6, k = i & 63;
        WgS[n * 72 + k] = WghT[i];
        WaS[n * 72 + k] = WahT[i];
    }

    // init state: h0 = tanh(s), n=d=0, amax=-1e38
    float nS[32], dS[32], am[32];
#pragma unroll
    for (int p = 0; p < 32; ++p) { nS[p] = 0.0f; dS[p] = 0.0f; am[p] = -1e38f; }
#pragma unroll
    for (int nt = 0; nt < 4; ++nt) {
        const int n = nt * 16 + ln;
        const float hv = tanh_fast(s[n]);
#pragma unroll
        for (int v = 0; v < 8; ++v) {
            const int m = v + 8 * h;
            hA[m * 66 + n] = (__bf16)hv;
        }
    }
    __syncthreads();

    for (int t = 0; t < SEQT; ++t) {
        // --- stage this step's [u|gx|ax] rows (16 x 192 f32, coalesced) ---
        for (int i = 0; i < 16; ++i) {
            const float* __restrict__ src =
                Out + ((long)input * BT + (long)(b0 + i) * SEQT + t) * NC;
#pragma unroll
            for (int j = 0; j < 6; ++j)
                xrow[i * 196 + j * 32 + lane] = src[j * 32 + lane];
        }
        __syncthreads();

        // --- A fragments from hA (K chunks 0..31, 32..63) ---
        Frag16 a0, a1;
        {
            const u32* Hd = (const u32*)hA;
            const int rb = ln * 33;
#pragma unroll
            for (int j = 0; j < 8; ++j) {
                a0.u[j] = Hd[rb + a_dw(j, h)];
                a1.u[j] = Hd[rb + 16 + a_dw(j, h)];
            }
        }

        // --- Gh and Ah: 4 N-tiles each, 2 WMMAs per tile (K=64) ---
        v8f accG[4], accA[4];
#pragma unroll
        for (int nt = 0; nt < 4; ++nt) {
            const __bf16* wgr = WgS + (nt * 16 + ln) * 72 + 16 * h;
            const __bf16* war = WaS + (nt * 16 + ln) * 72 + 16 * h;
            Frag16 bg0, bg1, ba0, ba1;
            bg0.f4[0] = ((const f32x4*)wgr)[0];
            bg0.f4[1] = ((const f32x4*)wgr)[1];
            bg1.f4[0] = ((const f32x4*)(wgr + 32))[0];
            bg1.f4[1] = ((const f32x4*)(wgr + 32))[1];
            ba0.f4[0] = ((const f32x4*)war)[0];
            ba0.f4[1] = ((const f32x4*)war)[1];
            ba1.f4[0] = ((const f32x4*)(war + 32))[0];
            ba1.f4[1] = ((const f32x4*)(war + 32))[1];

            v8f cg = v8f_zero();
            cg = __builtin_amdgcn_wmma_f32_16x16x32_bf16(
                false, a0.v, false, bg0.v, (short)0, cg, false, false);
            cg = __builtin_amdgcn_wmma_f32_16x16x32_bf16(
                false, a1.v, false, bg1.v, (short)0, cg, false, false);
            accG[nt] = cg;

            v8f ca = v8f_zero();
            ca = __builtin_amdgcn_wmma_f32_16x16x32_bf16(
                false, a0.v, false, ba0.v, (short)0, ca, false, false);
            ca = __builtin_amdgcn_wmma_f32_16x16x32_bf16(
                false, a1.v, false, ba1.v, (short)0, ca, false, false);
            accA[nt] = ca;
        }

        // --- stable RWA update; each lane owns its 32 (b,h) cells ---
#pragma unroll
        for (int nt = 0; nt < 4; ++nt) {
#pragma unroll
            for (int v = 0; v < 8; ++v) {
                const int p = nt * 8 + v;
                const int m = v + 8 * h;
                const int n = nt * 16 + ln;
                const float u  = xrow[m * 196 + n];
                const float gx = xrow[m * 196 + 64 + n];
                const float ax = xrow[m * 196 + 128 + n];
                const float g  = gx + accG[nt][v];
                const float a  = ax + accA[nt][v];
                const float z  = u * tanh_fast(g);
                const float anew = fmaxf(am[p], a);
                const float eo = __expf(am[p] - anew);
                const float ea = __expf(a - anew);
                nS[p] = nS[p] * eo + z * ea;
                dS[p] = dS[p] * eo + ea;
                const float hf = tanh_fast(nS[p] / dS[p]);
                am[p] = anew;
                hA[m * 66 + n] = (__bf16)hf;
                if (t == SEQT - 1)
                    hn[(long)(b0 + m) * 128 + input * 64 + n] = hf;
            }
        }
        __syncthreads();
    }
}

// ---------------------------------------------------------------------------
// Kernel H: batch-stat BN (training mode, gamma=1, beta=0) + W1/PReLU/W2 +
// sigmoid. 4 MFLOP total -> one 256-thread block, VALU.
// ---------------------------------------------------------------------------
__global__ __launch_bounds__(256) void rwa_head_kernel(
    const float* __restrict__ hn, const float* __restrict__ W1,
    const float* __restrict__ b1, const float* __restrict__ pw_,
    const float* __restrict__ W2, const float* __restrict__ b2_,
    float* __restrict__ out)
{
    __shared__ float W1s[128 * 64];
    __shared__ float meanS[128];
    __shared__ float rstdS[128];

    const int tid = threadIdx.x;
    for (int i = tid; i < 128 * 64; i += 256) W1s[i] = W1[i];

    if (tid < 128) {
        float s1 = 0.0f, s2 = 0.0f;
        for (int r = 0; r < BATCH; ++r) {
            const float v = hn[r * 128 + tid];
            s1 += v; s2 += v * v;
        }
        const float mean = s1 * (1.0f / (float)BATCH);
        const float var  = s2 * (1.0f / (float)BATCH) - mean * mean;
        meanS[tid] = mean;
        rstdS[tid] = rsqrtf(var + 1e-5f);
    }
    __syncthreads();

    const float pw = pw_[0];
    const float b2 = b2_[0];
    for (int r = tid; r < BATCH; r += 256) {
        float acc[64];
#pragma unroll
        for (int j = 0; j < 64; ++j) acc[j] = b1[j];
        for (int c = 0; c < 128; ++c) {
            const float nh = (hn[r * 128 + c] - meanS[c]) * rstdS[c];
#pragma unroll
            for (int j = 0; j < 64; ++j) acc[j] += nh * W1s[c * 64 + j];
        }
        float lin = b2;
#pragma unroll
        for (int j = 0; j < 64; ++j) {
            float aj = acc[j];
            aj = (aj >= 0.0f) ? aj : pw * aj;
            lin += aj * W2[j];
        }
        out[r] = 1.0f / (1.0f + __expf(-lin));
    }
}

// ---------------------------------------------------------------------------
// Host launcher
// ---------------------------------------------------------------------------
extern "C" void kernel_launch(void* const* d_in, const int* in_sizes, int n_in,
                              void* d_out, int out_size, void* d_ws, size_t ws_size,
                              hipStream_t stream)
{
    (void)in_sizes; (void)n_in; (void)out_size; (void)ws_size;

    const float* in1 = (const float*)d_in[0];
    const float* in2 = (const float*)d_in[1];
    const float* Wu  = (const float*)d_in[2];
    const float* bu  = (const float*)d_in[3];
    const float* Wg  = (const float*)d_in[4];
    const float* bg  = (const float*)d_in[5];
    const float* Wa  = (const float*)d_in[6];
    const float* s   = (const float*)d_in[7];
    const float* W1  = (const float*)d_in[8];
    const float* b1  = (const float*)d_in[9];
    const float* pw  = (const float*)d_in[10];
    const float* W2  = (const float*)d_in[11];
    const float* b2  = (const float*)d_in[12];
    float* out = (float*)d_out;

    // workspace layout (256B-aligned offsets)
    char* w = (char*)d_ws;
    __bf16* WcT  = (__bf16*)(w);                        // 192*320*2 = 122880
    __bf16* WghT = (__bf16*)(w + 122880);               // 8192
    __bf16* WahT = (__bf16*)(w + 131072);               // 8192
    float*  bias = (float*)(w + 139264);                // 768 (pad to 1024)
    float*  Out  = (float*)(w + 140288);                // 2*BT*192*4 = 100663296
    float*  hn   = (float*)(w + 140288 + 100663296ull); // 512*128*4 = 262144

    // 1) pack weights to bf16 fragment-friendly layouts
    {
        const int total = NC * KPAD + 8192 + NC;
        rwa_pack_kernel<<<(total + 255) / 256, 256, 0, stream>>>(
            Wu, bu, Wg, bg, Wa, WcT, WghT, WahT, bias);
    }
    // 2) time-parallel fused GEMM (TDM-staged): Out = [x1;x2] @ [Wu|Wg_x|Wa_x] + b
    rwa_gemm_kernel<<<1024, 256, 128 * D_IN * 4, stream>>>(in1, in2, WcT, bias, Out);
    // 3) sequential RWA scan (2 inputs x 32 batch strips)
    rwa_scan_kernel<<<64, 32, 0, stream>>>(Out, WghT, WahT, s, hn);
    // 4) BN + MLP head + sigmoid
    rwa_head_kernel<<<1, 256, 0, stream>>>(hn, W1, b1, pw, W2, b2, out);
}
